// NaiveDecoder_52115133169750
// MI455X (gfx1250) — matrix operand order repaired and checked by reading, not compiled, not run on hardware
//
#include <hip/hip_runtime.h>

// ---------- CDNA5 wave32 WMMA types ----------
typedef __attribute__((ext_vector_type(16))) __bf16   v16bf;
typedef __attribute__((ext_vector_type(8)))  float    v8f;
typedef __attribute__((ext_vector_type(4)))  unsigned u32x4;

#define HDIM   1024
#define EDIM   512
#define NBATCH 64
#define LSEQ   512
#define TSTEPS 511   // L-1
#define VOCAB  32000

#if defined(__has_builtin) && __has_builtin(__builtin_amdgcn_sched_group_barrier)
#define SCHED_WMMA(n)  __builtin_amdgcn_sched_group_barrier(0x008, (n), 0)  // MFMA/WMMA group
#define SCHED_VMEMR(n) __builtin_amdgcn_sched_group_barrier(0x020, (n), 0)  // VMEM read group
#else
#define SCHED_WMMA(n)
#define SCHED_VMEMR(n)
#endif

union Frag {
  v16bf    v;
  u32x4    q[2];
};

__device__ inline unsigned bf16r(float x) {           // f32 -> bf16 bits, RNE
  unsigned u = __float_as_uint(x);
  return (u + 0x7FFFu + ((u >> 16) & 1u)) >> 16;
}
__device__ inline float sigm(float x) { return 1.0f / (1.0f + __expf(-x)); }

// A-fragment (16x32, 16-bit) from a row pointer:
// lane half h: VGPR0-3 = K 8h..8h+7, VGPR4-7 = K 16+8h..23+8h
__device__ inline void load_a_row(Frag& f, const unsigned short* row, int kc, int h) {
  const u32x4* p = (const u32x4*)(row + kc * 32 + h * 8);
  f.q[0] = p[0];          // K = kc*32 + 8h   .. +7
  f.q[1] = p[2];          // K = kc*32 +16+8h .. +7
}
// B-fragment (32x16, col-major per lane): lane half h holds K 8h..8h+7, 16+8h..23+8h
__device__ inline void load_b(Frag& f, const unsigned short* wrow, int kc, int h) {
  const u32x4* p = (const u32x4*)(wrow + kc * 32 + h * 8);
  f.q[0] = p[0];
  f.q[1] = p[2];
}
__device__ inline v8f wmma_bf16(const Frag& a, const Frag& b, v8f c) {
  return __builtin_amdgcn_wmma_f32_16x16x32_bf16(false, a.v, false, b.v, (short)0, c, false, false);
}

// One pipeline stage: A fragment + 3 gate weight fragments -> 3 WMMAs
struct GSet { Frag a, r, z, n; };

__device__ inline void gload(GSet& s, const unsigned short* arow,
                             const unsigned short* wr, const unsigned short* wz,
                             const unsigned short* wn, int kc, int h) {
  load_a_row(s.a, arow, kc, h);
  load_b(s.r, wr, kc, h);
  load_b(s.z, wz, kc, h);
  load_b(s.n, wn, kc, h);
}
__device__ inline void gmac(const GSet& s, v8f& r, v8f& z, v8f& n) {
  r = wmma_bf16(s.a, s.r, r);
  z = wmma_bf16(s.a, s.z, z);
  n = wmma_bf16(s.a, s.n, n);
}

// Software-pipelined 3-gate GEMM over KC chunks of 32.
// Prefetch depth 2; sched_group_barrier pins a per-iteration pattern of
// {3 WMMA, 8 VMEM-loads} x2 so each buffer's loads fly under the other
// buffer's WMMAs (waits become loadcnt<=8 instead of full drains).
template<int KC>
__device__ inline void gemm3(const unsigned short* arow,
                             const unsigned short* wr, const unsigned short* wz,
                             const unsigned short* wn, int h,
                             v8f& accr, v8f& accz, v8f& accn) {
  GSet s0, s1;
  gload(s0, arow, wr, wz, wn, 0, h);
  if (KC > 1) gload(s1, arow, wr, wz, wn, 1, h);
#pragma unroll
  for (int kc = 0; kc < KC; kc += 2) {
    gmac(s0, accr, accz, accn);                             // consume chunk kc
    if (kc + 2 < KC) gload(s0, arow, wr, wz, wn, kc + 2, h);
    if (kc + 1 < KC) gmac(s1, accr, accz, accn);            // consume chunk kc+1
    if (kc + 3 < KC) gload(s1, arow, wr, wz, wn, kc + 3, h);
    SCHED_WMMA(3);
    SCHED_VMEMR(8);
    SCHED_WMMA(3);
    SCHED_VMEMR(8);
  }
}

// ---------- fused GRU cell step ----------
// KX: K-chunks of the input GEMM (16 for layer 0, 32 for layer 1)
// GATHER: x rows gathered via tokens from bf16 embedding table
// WOUT: also store h_new into dec_hids[m, t, :]
template<int KX, bool GATHER, bool WOUT>
__global__ void __launch_bounds__(256) gru_step(
    const int* __restrict__ tokens, const unsigned short* __restrict__ xsrc,
    const unsigned short* __restrict__ Wi, const unsigned short* __restrict__ Wh,
    const float* __restrict__ bi, const float* __restrict__ bh,
    const unsigned short* __restrict__ hprev_b, const float* __restrict__ hprev_f,
    unsigned short* __restrict__ hnew_b, float* __restrict__ hnew_f,
    float* __restrict__ dout, int t)
{
  const int lane = threadIdx.x & 31, w = threadIdx.x >> 5;
  const int h = lane >> 4, ln = lane & 15;
  const int nt = w & 1, mt = w >> 1;
  const int j = blockIdx.x * 32 + nt * 16 + ln;       // output column of H
  const int KXDIM = KX * 32;                          // input width (512 or 1024)

  const unsigned short* wi_r = Wi + (size_t)(0 * HDIM + j) * KXDIM;
  const unsigned short* wi_z = Wi + (size_t)(1 * HDIM + j) * KXDIM;
  const unsigned short* wi_n = Wi + (size_t)(2 * HDIM + j) * KXDIM;
  const unsigned short* wh_r = Wh + (size_t)(0 * HDIM + j) * HDIM;
  const unsigned short* wh_z = Wh + (size_t)(1 * HDIM + j) * HDIM;
  const unsigned short* wh_n = Wh + (size_t)(2 * HDIM + j) * HDIM;

  const int m_a = mt * 16 + ln;                       // A-matrix row (batch)
  const unsigned short* xrow;
  if (GATHER) {
    const int tok = tokens[m_a * LSEQ + t];
    xrow = xsrc + (size_t)tok * KXDIM;                // bf16 embedding row
  } else {
    xrow = xsrc + (size_t)m_a * KXDIM;                // h0_new row
  }
  const unsigned short* hrow = hprev_b + (size_t)m_a * HDIM;

  v8f ar = {}, az = {}, ain = {}, ahn = {};

  // gi = x @ W_ih^T  (r,z combined later; n-part kept separate in ain)
  gemm3<KX>(xrow, wi_r, wi_z, wi_n, h, ar, az, ain);
  // gh = h @ W_hh^T  (r,z accumulate on top; n-part separate in ahn)
  gemm3<HDIM / 32>(hrow, wh_r, wh_z, wh_n, h, ar, az, ahn);

  const float bir = bi[j], biz = bi[HDIM + j], bin = bi[2 * HDIM + j];
  const float bhr = bh[j], bhz = bh[HDIM + j], bhn = bh[2 * HDIM + j];
#pragma unroll
  for (int rr = 0; rr < 8; ++rr) {
    const int m = mt * 16 + rr + 8 * h;               // C/D layout: m = vgpr + 8*(lane>>4)
    const float hp = hprev_f[m * HDIM + j];
    const float r  = sigm(ar[rr] + bir + bhr);
    const float z  = sigm(az[rr] + biz + bhz);
    const float nn = tanhf(ain[rr] + bin + r * (ahn[rr] + bhn));
    const float hv = (1.0f - z) * nn + z * hp;
    hnew_f[m * HDIM + j] = hv;
    hnew_b[m * HDIM + j] = (unsigned short)bf16r(hv);
    if (WOUT)
      dout[(size_t)m * (TSTEPS * HDIM) + (size_t)t * HDIM + j] = hv;  // dec_hids[m,t,:]
  }
}

// ---------- prep / tail kernels ----------
__global__ void cvt_bf16(const float* __restrict__ s, unsigned short* __restrict__ d, int n) {
  int i = blockIdx.x * blockDim.x + threadIdx.x;
  if (i < n) d[i] = (unsigned short)bf16r(s[i]);
}
__global__ void init_h(const float* __restrict__ dec_init,
                       float* h0f, unsigned short* h0b, float* h1f, unsigned short* h1b) {
  int i = blockIdx.x * blockDim.x + threadIdx.x;
  if (i < NBATCH * HDIM) {
    float a = dec_init[i];
    float b = dec_init[NBATCH * HDIM + i];
    h0f[i] = a; h0b[i] = (unsigned short)bf16r(a);
    h1f[i] = b; h1b[i] = (unsigned short)bf16r(b);
  }
}
__global__ void finalize(const float* __restrict__ h0f, const float* __restrict__ h1f,
                         float* __restrict__ tail) {
  int i = blockIdx.x * blockDim.x + threadIdx.x;
  if (i < NBATCH * HDIM) {
    tail[i] = h0f[i];
    tail[NBATCH * HDIM + i] = h1f[i];
  }
}

extern "C" void kernel_launch(void* const* d_in, const int* in_sizes, int n_in,
                              void* d_out, int out_size, void* d_ws, size_t ws_size,
                              hipStream_t stream) {
  const int*   tokens   = (const int*)  d_in[0];
  const float* dec_init = (const float*)d_in[1];
  const float* emb      = (const float*)d_in[2];
  const float* W_ih0    = (const float*)d_in[3];
  const float* W_hh0    = (const float*)d_in[4];
  const float* b_ih0    = (const float*)d_in[5];
  const float* b_hh0    = (const float*)d_in[6];
  const float* W_ih1    = (const float*)d_in[7];
  const float* W_hh1    = (const float*)d_in[8];
  const float* b_ih1    = (const float*)d_in[9];
  const float* b_hh1    = (const float*)d_in[10];
  float* dout = (float*)d_out;

  char* ws = (char*)d_ws;
  auto carve = [&](size_t bytes) -> void* {
    void* p = (void*)ws;
    ws += (bytes + 255) & ~(size_t)255;
    return p;
  };
  const int nWi0 = 3 * HDIM * EDIM;      // 1.57M
  const int nWh  = 3 * HDIM * HDIM;      // 3.15M
  const int nEmb = VOCAB * EDIM;         // 16.4M
  unsigned short* Wih0b = (unsigned short*)carve((size_t)nWi0 * 2);
  unsigned short* Whh0b = (unsigned short*)carve((size_t)nWh * 2);
  unsigned short* Wih1b = (unsigned short*)carve((size_t)nWh * 2);
  unsigned short* Whh1b = (unsigned short*)carve((size_t)nWh * 2);
  unsigned short* embb  = (unsigned short*)carve((size_t)nEmb * 2);
  unsigned short* h0b[2] = {(unsigned short*)carve(NBATCH * HDIM * 2),
                            (unsigned short*)carve(NBATCH * HDIM * 2)};
  unsigned short* h1b[2] = {(unsigned short*)carve(NBATCH * HDIM * 2),
                            (unsigned short*)carve(NBATCH * HDIM * 2)};
  float* h0f[2] = {(float*)carve(NBATCH * HDIM * 4), (float*)carve(NBATCH * HDIM * 4)};
  float* h1f[2] = {(float*)carve(NBATCH * HDIM * 4), (float*)carve(NBATCH * HDIM * 4)};

  // one-time: weights + embedding table -> bf16 (persist in 192MB L2), init hidden state
  cvt_bf16<<<(nWi0 + 255) / 256, 256, 0, stream>>>(W_ih0, Wih0b, nWi0);
  cvt_bf16<<<(nWh + 255) / 256, 256, 0, stream>>>(W_hh0, Whh0b, nWh);
  cvt_bf16<<<(nWh + 255) / 256, 256, 0, stream>>>(W_ih1, Wih1b, nWh);
  cvt_bf16<<<(nWh + 255) / 256, 256, 0, stream>>>(W_hh1, Whh1b, nWh);
  cvt_bf16<<<(nEmb + 255) / 256, 256, 0, stream>>>(emb, embb, nEmb);
  init_h<<<(NBATCH * HDIM + 255) / 256, 256, 0, stream>>>(dec_init, h0f[0], h0b[0], h1f[0], h1b[0]);

  // sequential scan: 2 fused WMMA kernels per timestep, hidden state ping-pong
  for (int t = 0; t < TSTEPS; ++t) {
    const int p = t & 1, q = p ^ 1;
    gru_step<EDIM / 32, true, false><<<HDIM / 32, 256, 0, stream>>>(
        tokens, embb, Wih0b, Whh0b, b_ih0, b_hh0,
        h0b[p], h0f[p], h0b[q], h0f[q], nullptr, t);
    gru_step<HDIM / 32, false, true><<<HDIM / 32, 256, 0, stream>>>(
        nullptr, h0b[q], Wih1b, Whh1b, b_ih1, b_hh1,
        h1b[p], h1f[p], h1b[q], h1f[q], dout, t);
  }

  // dec_prev = stack(h0_final, h1_final); final buffers are index (TSTEPS & 1) = 1
  finalize<<<(NBATCH * HDIM + 255) / 256, 256, 0, stream>>>(
      h0f[1], h1f[1], dout + (size_t)NBATCH * TSTEPS * HDIM);
}
